// RNN_2078764171663
// MI455X (gfx1250) — matrix operand order repaired and checked by reading, not compiled, not run on hardware
//
#include <hip/hip_runtime.h>
#include <hip/hip_bf16.h>

// Elman RNN forward for MI455X (gfx1250, wave32, WMMA).
//   V=50257, H=512, O=4, B=64, T=512
// Pipeline:
//   k0: swizzle W_in / W_h (f32 [N][K]) -> bf16 B-fragment layout
//   k1: A = emb[ids] @ W_in^T + b_h, stored in C-fragment order (WMMA bf16)
//   k2: per-batch-tile recurrent scan h = relu(A_t + h @ W_h^T)  (WMMA,
//       W_h held in registers across all timesteps, A pipelined across t,
//       bank-conflict-free LDS h double buffer)
//   k3: head + softmax + log_softmax + NLL loss

#define Hh   512
#define Bsz  64
#define Tlen 512
#define Osz  4
#define SKEW 8
#define HP   (Hh + SKEW)    // padded LDS row stride (halfs): rows 4 banks apart

typedef __attribute__((ext_vector_type(16))) __bf16 v16bf;
typedef __attribute__((ext_vector_type(8)))  __bf16 v8bf;
typedef __attribute__((ext_vector_type(8)))  float  v8f;

union BF16Frag { v16bf v; v8bf h8[2]; };

__device__ __forceinline__ unsigned short bf16_rne(float f) {
    union { float f; unsigned int u; } c; c.f = f;
    unsigned int u = c.u;
    return (unsigned short)((u + 0x7FFFu + ((u >> 16) & 1u)) >> 16);
}

// ---------------------------------------------------------------------------
// k0: pre-swizzle weights into WMMA B-fragment order.
// idx = blk*512 + lane*16 + j, blk = ntile*16 + ktile, value = W[n][k] with
//   n = ntile*16 + (lane&15),  k = ktile*32 + (lane>>4)*8 + (j&7) + (j>>3)*16.
// Each lane then fetches its whole fragment as 32 contiguous bytes.
// ---------------------------------------------------------------------------
__global__ __launch_bounds__(256) void swizzle_kernel(
    const float* __restrict__ Win, const float* __restrict__ Wh,
    unsigned short* __restrict__ SwzIn, unsigned short* __restrict__ SwzH)
{
    int gidx = blockIdx.x * 256 + threadIdx.x;   // 0 .. 2*262144-1
    const int mat = gidx >> 18;
    const int idx = gidx & 262143;
    const int j     = idx & 15;
    const int lane  = (idx >> 4) & 31;
    const int blk   = idx >> 9;                  // 0..511
    const int ntile = blk >> 4, ktile = blk & 15;
    const int n = ntile * 16 + (lane & 15);
    const int k = ktile * 32 + (lane >> 4) * 8 + (j & 7) + (j >> 3) * 16;
    const float* src = (mat == 0) ? Win : Wh;
    unsigned short* dst = (mat == 0) ? SwzIn : SwzH;
    dst[idx] = bf16_rne(src[(size_t)n * Hh + k]);
}

// ---------------------------------------------------------------------------
// k1: A(t,b,n) = sum_k emb[ids[b][t]][k] * W_in[n][k] + b_h[n]
// One workgroup (8 waves) per 16-row (t,btile) tile; each wave owns 4
// N-tiles.  Output stored in C-fragment order:
//   Aswz[((t*4 + btile)*32 + ntile)*256 + lane*8 + r]
// so the scan kernel reads each accumulator as one contiguous v8f.
// ---------------------------------------------------------------------------
__global__ __launch_bounds__(256) void embed_gemm_kernel(
    const int*   __restrict__ ids,    // [B][T]
    const float* __restrict__ emb,    // [V][H]
    const unsigned short* __restrict__ Wswz,  // W_in swizzled
    const float* __restrict__ b_h,    // [H]
    float*       __restrict__ Aswz)   // [T][4][32][32][8]
{
    __shared__ unsigned short xs[16 * HP];           // padded bf16 staging
    const int tid   = threadIdx.x;
    const int Mtile = blockIdx.x;                    // 0..2047
    const int t     = Mtile >> 2;
    const int btile = Mtile & 3;
    const int bbase = btile * 16;

    for (int i = tid; i < 16 * Hh; i += 256) {
        const int row = i >> 9, col = i & 511;
        const int id  = ids[(bbase + row) * Tlen + t];
        xs[row * HP + col] = bf16_rne(emb[(size_t)id * Hh + col]);
    }
    __syncthreads();

    const int lane = tid & 31, wave = tid >> 5;
    const int half = lane >> 4, l16 = lane & 15;
    const int nt0  = wave * 4;
    const unsigned short* hrow  = &xs[l16 * HP + half * 8];
    const unsigned short* wbase = Wswz + (size_t)nt0 * (16 * 512) + lane * 16;

    v8f acc[4];
#pragma unroll
    for (int nt = 0; nt < 4; ++nt) {
        const float bv = b_h[(nt0 + nt) * 16 + l16];
#pragma unroll
        for (int r = 0; r < 8; ++r) acc[nt][r] = bv;
    }

    // software-pipelined K loop (fully unrolled: buffer swap = SSA rename)
    BF16Frag af; v16bf bf[4];
    af.h8[0] = *(const v8bf*)(hrow);
    af.h8[1] = *(const v8bf*)(hrow + 16);
#pragma unroll
    for (int nt = 0; nt < 4; ++nt)
        bf[nt] = *(const v16bf*)(wbase + ((nt * 16) << 9));

#pragma unroll
    for (int ks = 0; ks < 15; ++ks) {
        BF16Frag afn; v16bf bfn[4];
        afn.h8[0] = *(const v8bf*)(hrow + (ks + 1) * 32);
        afn.h8[1] = *(const v8bf*)(hrow + (ks + 1) * 32 + 16);
#pragma unroll
        for (int nt = 0; nt < 4; ++nt)
            bfn[nt] = *(const v16bf*)(wbase + ((nt * 16 + ks + 1) << 9));
#pragma unroll
        for (int nt = 0; nt < 4; ++nt)
            acc[nt] = __builtin_amdgcn_wmma_f32_16x16x32_bf16(
                false, af.v, false, bf[nt], (short)0, acc[nt], false, false);
        af.v = afn.v;
#pragma unroll
        for (int nt = 0; nt < 4; ++nt) bf[nt] = bfn[nt];
    }
#pragma unroll
    for (int nt = 0; nt < 4; ++nt)
        acc[nt] = __builtin_amdgcn_wmma_f32_16x16x32_bf16(
            false, af.v, false, bf[nt], (short)0, acc[nt], false, false);

    const size_t obase = ((size_t)t * 4 + btile) * (32 * 256) + lane * 8;
#pragma unroll
    for (int nt = 0; nt < 4; ++nt)
        *(v8f*)&Aswz[obase + (size_t)(nt0 + nt) * 256] = acc[nt];
}

// ---------------------------------------------------------------------------
// k2: recurrent scan.  Batch rows are independent: 4 workgroups each own a
// 16-row batch tile and loop t privately (no cross-WG sync).
//  - W_h B-fragments are loop-invariant across t: each wave holds its 32
//    fragments (2 N-tiles x 16 k-steps) in 256 VGPRs for the whole scan,
//    so the steady-state K loop has NO global loads at all.
//  - A accumulators for step t+1 are loaded at the top of step t (latency
//    hidden behind 32 WMMAs + barrier).
//  - h double-buffered in LDS as bf16 with 4-bank row skew (conflict-free
//    ds_load_b128 fragments); A-fragment ds loads pipelined one k ahead.
// ---------------------------------------------------------------------------
__global__ __launch_bounds__(512) void rnn_scan_kernel(
    const float* __restrict__ Aswz,           // [T][4][32][32][8] fragment order
    const unsigned short* __restrict__ Wswz,  // W_h swizzled
    float* __restrict__ hT)                   // [B][H] row-major
{
    __shared__ unsigned short hbuf[2][16 * HP];
    const int tid  = threadIdx.x;
    const int lane = tid & 31, wave = tid >> 5;      // wave 0..15
    const int half = lane >> 4, l16 = lane & 15;
    const int btile = blockIdx.x;                    // 0..3
    const int mbase = btile * 16;
    const int nt0   = wave * 2;
    const int arow  = l16 * HP + half * 8;

    // W_h fragments, resident in registers for the entire scan.
    const unsigned short* wbase = Wswz + (size_t)nt0 * (16 * 512) + lane * 16;
    v16bf bW[32];                                    // [nt*16 + ks]
#pragma unroll
    for (int i = 0; i < 32; ++i)
        bW[i] = *(const v16bf*)(wbase + (i << 9));

    for (int i = tid; i < 16 * HP; i += 512) hbuf[0][i] = 0;   // h0 = 0
    __syncthreads();

    // A accumulators for t=0 (pipelined one step ahead thereafter)
    v8f accp[2];
#pragma unroll
    for (int nt = 0; nt < 2; ++nt)
        accp[nt] = *(const v8f*)&Aswz[(size_t)btile * (32 * 256) +
                                      (size_t)(nt0 + nt) * 256 + lane * 8];

    for (int t = 0; t < Tlen; ++t) {
        const int cur = t & 1, nxt = cur ^ 1;

        v8f acc[2];
#pragma unroll
        for (int nt = 0; nt < 2; ++nt) acc[nt] = accp[nt];

        if (t + 1 < Tlen) {   // issue next step's A loads now; consumed next iter
            const size_t abase = ((size_t)(t + 1) * 4 + btile) * (32 * 256) + lane * 8;
#pragma unroll
            for (int nt = 0; nt < 2; ++nt)
                accp[nt] = *(const v8f*)&Aswz[abase + (size_t)(nt0 + nt) * 256];
        }

        const unsigned short* hrow = &hbuf[cur][arow];
        BF16Frag af;
        af.h8[0] = *(const v8bf*)(hrow);
        af.h8[1] = *(const v8bf*)(hrow + 16);
#pragma unroll
        for (int ks = 0; ks < 15; ++ks) {
            BF16Frag afn;
            afn.h8[0] = *(const v8bf*)(hrow + (ks + 1) * 32);
            afn.h8[1] = *(const v8bf*)(hrow + (ks + 1) * 32 + 16);
#pragma unroll
            for (int nt = 0; nt < 2; ++nt)
                acc[nt] = __builtin_amdgcn_wmma_f32_16x16x32_bf16(
                    false, af.v, false, bW[nt * 16 + ks], (short)0, acc[nt], false, false);
            af.v = afn.v;
        }
#pragma unroll
        for (int nt = 0; nt < 2; ++nt)
            acc[nt] = __builtin_amdgcn_wmma_f32_16x16x32_bf16(
                false, af.v, false, bW[nt * 16 + 15], (short)0, acc[nt], false, false);

        // relu + write next-step h (and final h_T)
#pragma unroll
        for (int nt = 0; nt < 2; ++nt) {
            const int n = (nt0 + nt) * 16 + l16;
#pragma unroll
            for (int r = 0; r < 8; ++r) {
                float v = acc[nt][r];
                v = v > 0.f ? v : 0.f;
                const int m = r + 8 * half;
                hbuf[nxt][m * HP + n] = bf16_rne(v);
                if (t == Tlen - 1) hT[(size_t)(mbase + m) * Hh + n] = v;
            }
        }
        __syncthreads();
    }
}

// ---------------------------------------------------------------------------
// k3: logits = softmax(hT @ W_out^T + b_out); logp = log_softmax(logits)
// (faithful softmax-before-CE bug); loss = -mean_b logp[b, label[b]].
// out[0] = loss, out[1..256] = logits row-major.  One block, deterministic.
// ---------------------------------------------------------------------------
__global__ __launch_bounds__(64) void head_kernel(
    const float* __restrict__ hT,     // [B][H]
    const float* __restrict__ W_out,  // [O][H]
    const float* __restrict__ b_out,  // [O]
    const int*   __restrict__ labels, // [B]
    float* __restrict__ out)          // [1 + B*O]
{
    __shared__ float lsum[Bsz];
    const int b = threadIdx.x;

    float z[Osz];
#pragma unroll
    for (int o = 0; o < Osz; ++o) {
        float s = b_out[o];
        for (int k = 0; k < Hh; ++k) s += hT[(size_t)b * Hh + k] * W_out[o * Hh + k];
        z[o] = s;
    }
    float m1 = z[0];
#pragma unroll
    for (int o = 1; o < Osz; ++o) m1 = fmaxf(m1, z[o]);
    float p[Osz], s1 = 0.f;
#pragma unroll
    for (int o = 0; o < Osz; ++o) { p[o] = __expf(z[o] - m1); s1 += p[o]; }
    const float inv = 1.f / s1;
#pragma unroll
    for (int o = 0; o < Osz; ++o) p[o] *= inv;
    float m2 = p[0];
#pragma unroll
    for (int o = 1; o < Osz; ++o) m2 = fmaxf(m2, p[o]);
    float s2 = 0.f;
#pragma unroll
    for (int o = 0; o < Osz; ++o) s2 += __expf(p[o] - m2);
    const float lse = m2 + __logf(s2);

#pragma unroll
    for (int o = 0; o < Osz; ++o) out[1 + b * Osz + o] = p[o];
    const int lab = labels[b];
    lsum[b] = -(p[lab] - lse);
    __syncthreads();
    if (b == 0) {
        float s = 0.f;
        for (int i = 0; i < Bsz; ++i) s += lsum[i];
        out[0] = s * (1.f / (float)Bsz);
    }
}

// ---------------------------------------------------------------------------
extern "C" void kernel_launch(void* const* d_in, const int* in_sizes, int n_in,
                              void* d_out, int out_size, void* d_ws, size_t ws_size,
                              hipStream_t stream) {
    const int*   ids    = (const int*)d_in[0];
    const int*   labels = (const int*)d_in[1];
    const float* emb    = (const float*)d_in[2];
    const float* W_in   = (const float*)d_in[3];
    const float* W_h    = (const float*)d_in[4];
    const float* b_h    = (const float*)d_in[5];
    const float* W_out  = (const float*)d_in[6];
    const float* b_out  = (const float*)d_in[7];

    // Workspace layout:
    //   [0, 512K)     W_in swizzled bf16
    //   [512K, 1M)    W_h  swizzled bf16
    //   [1M, 1M+64M)  Aswz[T][4][32][32][8] f32 (fragment order)
    //   [+128K)       hT[B][H] f32
    char* ws = (char*)d_ws;
    unsigned short* SwzIn = (unsigned short*)ws;
    unsigned short* SwzH  = (unsigned short*)(ws + (512u << 10));
    float* Aswz = (float*)(ws + (1u << 20));
    float* hT   = (float*)(ws + (1u << 20) + (size_t)Tlen * Bsz * Hh * sizeof(float));

    swizzle_kernel   <<<2048, 256, 0, stream>>>(W_in, W_h, SwzIn, SwzH);
    embed_gemm_kernel<<<2048, 256, 0, stream>>>(ids, emb, SwzIn, b_h, Aswz);
    rnn_scan_kernel  <<<4,    512, 0, stream>>>(Aswz, SwzH, hT);
    head_kernel      <<<1,    64,  0, stream>>>(hT, W_out, b_out, labels, (float*)d_out);
}
